// EncoderLSTM_32615981646087
// MI455X (gfx1250) — matrix operand order repaired and checked by reading, not compiled, not run on hardware
//
#include <hip/hip_runtime.h>
#include <hip/hip_bf16.h>
#include <cstdint>

// ---------------------------------------------------------------------------
// EncoderLSTM for MI455X (gfx1250, wave32, WMMA 16x16x32 bf16)
//   T=128, B=64, V=2048, H=1024, gates = 4H = 4096
// Strategy:
//   1) pack x, Wi, Wh to bf16 in WMMA per-lane fragment layouts
//   2) one big WMMA GEMM: gates_x = x @ Wi + b   (time-parallel)
//   3) persistent recurrent kernel: per step h @ Wh via WMMA, LSTM cell,
//      EOS masking, global barrier, double-buffered packed h
// ---------------------------------------------------------------------------

typedef __attribute__((ext_vector_type(16))) __bf16 v16bf;
typedef __attribute__((ext_vector_type(8)))  __bf16 v8bf;
typedef __attribute__((ext_vector_type(8)))  float  v8f;

#define T_STEPS 128
#define BATCH   64
#define VDIM    2048
#define HDIM    1024
#define NGATE   4096   // 4*H
#define NBLK    64     // persistent blocks; each owns 16 hidden columns

__device__ __forceinline__ __bf16 f2bf(float f) {
  union { float f; uint32_t u; } v; v.f = f;
  uint32_t r = v.u + 0x7FFFu + ((v.u >> 16) & 1u);   // round-to-nearest-even
  unsigned short h = (unsigned short)(r >> 16);
  __bf16 b;
  __builtin_memcpy(&b, &h, sizeof(h));
  return b;
}

__device__ __forceinline__ float sigf(float x) { return 1.0f / (1.0f + __expf(-x)); }

__device__ __forceinline__ v8f wmma_bf16(v16bf a, v16bf b, v8f c) {
  // (neg_a, A, neg_b, B, c_mod, C, reuse_a, reuse_b)
  return __builtin_amdgcn_wmma_f32_16x16x32_bf16(false, a, false, b, (short)0, c,
                                                 false, false);
}

// ---------------------------------------------------------------------------
// Fragment packing (ISA 7.12.2 layouts, wave32, 16-bit data)
// A (16x32 MxK): lane L holds row M=L%16; element e (0..15) maps
//   K = (e/8)*16 + (L/16)*8 + (e%8)
// B (32x16 KxN): lane L holds col N=L%16; element e maps
//   K = (e/8)*16 + (L/16)*8 + (e%8)
// Packed storage: [tile][ktile][lane][16 elems] -> each lane reads 32
// contiguous bytes per fragment.
// ---------------------------------------------------------------------------

__global__ __launch_bounds__(256) void pack_a_kernel(const float* __restrict__ src,
                                                     __bf16* __restrict__ dst,
                                                     int K, int KT, int n) {
  int d = blockIdx.x * 256 + threadIdx.x;
  if (d >= n) return;
  int e    = d & 15;
  int lane = (d >> 4) & 31;
  int kt   = (d >> 9) % KT;
  int mt   = d / (KT << 9);
  int row  = (mt << 4) + (lane & 15);
  int kk   = (kt << 5) + ((e >> 3) << 4) + ((lane >> 4) << 3) + (e & 7);
  dst[d] = f2bf(src[(size_t)row * K + kk]);
}

__global__ __launch_bounds__(256) void pack_b_kernel(const float* __restrict__ src,
                                                     __bf16* __restrict__ dst,
                                                     int N, int KT, int n) {
  int d = blockIdx.x * 256 + threadIdx.x;
  if (d >= n) return;
  int e    = d & 15;
  int lane = (d >> 4) & 31;
  int kt   = (d >> 9) % KT;
  int nt   = d / (KT << 9);
  int col  = (nt << 4) + (lane & 15);
  int kk   = (kt << 5) + ((e >> 3) << 4) + ((lane >> 4) << 3) + (e & 7);
  dst[d] = f2bf(src[(size_t)kk * N + col]);
}

// EOS prefix-scan: mask[t][b] = eos state BEFORE step t (OR of x[t'<t][b][1]!=0)
__global__ void eos_scan_kernel(const float* __restrict__ x,
                                const unsigned char* __restrict__ e0,
                                float* __restrict__ eos) {
  int b = threadIdx.x;
  if (b >= BATCH) return;
  float cur = (e0[b] != 0) ? 1.0f : 0.0f;
  for (int t = 0; t < T_STEPS; ++t) {
    eos[t * BATCH + b] = cur;
    if (x[((size_t)t * BATCH + b) * VDIM + 1] != 0.0f) cur = 1.0f;
  }
}

// ---------------------------------------------------------------------------
// gates_x = x @ Wi + b   (M=8192, N=4096, K=2048), fp32 accumulate/out.
// Block computes a 64x64 tile; 8 waves cover 16 (mt,nt) 16x16 sub-tiles,
// each wave handles 2 sub-tiles sharing the B fragment.
// ---------------------------------------------------------------------------
__global__ __launch_bounds__(256) void gemm_bias_kernel(const __bf16* __restrict__ Ap,
                                                        const __bf16* __restrict__ Bp,
                                                        const float* __restrict__ bias,
                                                        float* __restrict__ C,
                                                        int KT, int N) {
  const v16bf* A = (const v16bf*)Ap;
  const v16bf* B = (const v16bf*)Bp;
  int tid = threadIdx.x, wave = tid >> 5, lane = tid & 31;
  int gmt0 = blockIdx.x * 4 + (wave >> 2);   // 0..1 local
  int gmt1 = gmt0 + 2;                       // 2..3 local
  int gnt  = blockIdx.y * 4 + (wave & 3);
  const v16bf* Ar0 = A + (size_t)gmt0 * KT * 32 + lane;
  const v16bf* Ar1 = A + (size_t)gmt1 * KT * 32 + lane;
  const v16bf* Br  = B + (size_t)gnt  * KT * 32 + lane;
  v8f acc0 = {}, acc1 = {};
  for (int kt = 0; kt < KT; ++kt) {
    v16bf bf = Br[kt * 32];
    v16bf a0 = Ar0[kt * 32];
    v16bf a1 = Ar1[kt * 32];
    acc0 = wmma_bf16(a0, bf, acc0);
    acc1 = wmma_bf16(a1, bf, acc1);
  }
  int col = gnt * 16 + (lane & 15);
  float bv = bias[col];
#pragma unroll
  for (int r = 0; r < 8; ++r) {
    int row0 = gmt0 * 16 + r + ((lane >> 4) << 3);
    int row1 = gmt1 * 16 + r + ((lane >> 4) << 3);
    C[(size_t)row0 * N + col] = acc0[r] + bv;
    C[(size_t)row1 * N + col] = acc1[r] + bv;
  }
}

// Simple two-counter grid barrier (count + generation) in global memory.
__device__ __forceinline__ void grid_barrier(int* bar, int nblk) {
  __syncthreads();
  if (threadIdx.x == 0) {
    __threadfence();
    int g = __hip_atomic_load(&bar[1], __ATOMIC_ACQUIRE, __HIP_MEMORY_SCOPE_AGENT);
    int prev = __hip_atomic_fetch_add(&bar[0], 1, __ATOMIC_ACQ_REL, __HIP_MEMORY_SCOPE_AGENT);
    if (prev == nblk - 1) {
      __hip_atomic_store(&bar[0], 0, __ATOMIC_RELAXED, __HIP_MEMORY_SCOPE_AGENT);
      __hip_atomic_fetch_add(&bar[1], 1, __ATOMIC_RELEASE, __HIP_MEMORY_SCOPE_AGENT);
    } else {
      while (__hip_atomic_load(&bar[1], __ATOMIC_ACQUIRE, __HIP_MEMORY_SCOPE_AGENT) == g)
        __builtin_amdgcn_s_sleep(1);
    }
  }
  __syncthreads();
}

// ---------------------------------------------------------------------------
// Persistent recurrent kernel. Block w (0..63) owns hidden columns
// [16w, 16w+16). 8 waves: wave&3 = mtile (batch rows), wave>>2 selects gate
// pair {0,2} or {1,3} (shares the A fragment between two WMMA chains).
// hP holds two packed-A (64x1024) bf16 buffers, ping-ponged per step.
// ---------------------------------------------------------------------------
__global__ __launch_bounds__(256) void lstm_rec_kernel(const float* __restrict__ gates_x,
                                                       const __bf16* __restrict__ WhP,
                                                       __bf16* __restrict__ hP,
                                                       const float* __restrict__ eos,
                                                       const float* __restrict__ c0,
                                                       const float* __restrict__ h0,
                                                       float* __restrict__ out,
                                                       int* bar) {
  const int w = blockIdx.x;                 // hidden-column slice owner
  const int tid = threadIdx.x, wave = tid >> 5, lane = tid & 31;
  __shared__ float g_lds[4][BATCH][16];     // i,f,g,o tiles (fp32)
  __shared__ float h_lds[BATCH][16];        // masked h slice for repacking

  // per-thread LSTM state: 4 (row,col) cells of the (64 x 16) slice
  float c_reg[4], h_reg[4];
#pragma unroll
  for (int q = 0; q < 4; ++q) {
    int idx = tid + 256 * q, row = idx >> 4, col = idx & 15;
    c_reg[q] = c0[(size_t)row * HDIM + w * 16 + col];
    h_reg[q] = h0[(size_t)row * HDIM + w * 16 + col];
  }

  const v16bf* Wh = (const v16bf*)WhP;
  const int mt    = wave & 3;
  const int gate0 = wave >> 2;              // 0 or 1
  const int gate1 = gate0 + 2;              // 2 or 3
  const int HBUF  = BATCH * HDIM;           // elems per packed-h buffer
  const v16bf* B0 = Wh + (size_t)(gate0 * 64 + w) * 32 * 32 + lane;
  const v16bf* B1 = Wh + (size_t)(gate1 * 64 + w) * 32 * 32 + lane;

  for (int t = 0; t < T_STEPS; ++t) {
    const v16bf* hA = (const v16bf*)(hP + (size_t)(t & 1) * HBUF) + mt * 32 * 32 + lane;

    v8f acc0, acc1;                          // seed with x@Wi + b
#pragma unroll
    for (int r = 0; r < 8; ++r) {
      int row = mt * 16 + r + ((lane >> 4) << 3);
      int col = lane & 15;
      size_t base = ((size_t)(t * BATCH + row)) * NGATE + w * 16 + col;
      acc0[r] = gates_x[base + (size_t)gate0 * HDIM];
      acc1[r] = gates_x[base + (size_t)gate1 * HDIM];
    }
    for (int kt = 0; kt < 32; ++kt) {        // K = 1024 = 32 x 32
      v16bf a  = hA[kt * 32];
      v16bf b0 = B0[kt * 32];
      v16bf b1 = B1[kt * 32];
      acc0 = wmma_bf16(a, b0, acc0);
      acc1 = wmma_bf16(a, b1, acc1);
    }
#pragma unroll
    for (int r = 0; r < 8; ++r) {
      int row = mt * 16 + r + ((lane >> 4) << 3);
      int col = lane & 15;
      g_lds[gate0][row][col] = acc0[r];
      g_lds[gate1][row][col] = acc1[r];
    }
    __syncthreads();

    // LSTM cell + EOS masking; ys collects UNMASKED new_h (per reference)
#pragma unroll
    for (int q = 0; q < 4; ++q) {
      int idx = tid + 256 * q, row = idx >> 4, col = idx & 15;
      float gi = g_lds[0][row][col], gf = g_lds[1][row][col];
      float gg = g_lds[2][row][col], go = g_lds[3][row][col];
      float nc = sigf(gf) * c_reg[q] + sigf(gi) * tanhf(gg);
      float nh = sigf(go) * tanhf(nc);
      bool  m  = eos[t * BATCH + row] != 0.0f;
      c_reg[q] = m ? c_reg[q] : nc;
      float hn = m ? h_reg[q] : nh;
      h_reg[q] = hn;
      out[((size_t)(t * BATCH + row)) * HDIM + w * 16 + col] = nh;
      h_lds[row][col] = hn;
    }
    __syncthreads();

    // Repack masked h slice into next step's packed-A buffer.
    // kt = w/2; this WG's 16 columns are half (w&1) of that 32-wide K tile.
    if (wave < 4) {
      int pm = wave, kt = w >> 1, hh = w & 1;
      v8bf v;
#pragma unroll
      for (int e = 0; e < 8; ++e) {
        int srow = pm * 16 + (lane & 15);
        int ncol = ((lane >> 4) << 3) + e;
        v[e] = f2bf(h_lds[srow][ncol]);
      }
      __bf16* dst = hP + (size_t)((t + 1) & 1) * HBUF +
                    ((((size_t)(pm * 32 + kt)) * 32 + lane) << 4) + (hh << 3);
      *(v8bf*)dst = v;
    }
    grid_barrier(bar, NBLK);
  }
}

// ---------------------------------------------------------------------------
extern "C" void kernel_launch(void* const* d_in, const int* in_sizes, int n_in,
                              void* d_out, int out_size, void* d_ws, size_t ws_size,
                              hipStream_t stream) {
  (void)in_sizes; (void)n_in; (void)out_size; (void)ws_size;
  const float* x  = (const float*)d_in[0];   // (T,B,V)
  const float* Wi = (const float*)d_in[1];   // (V,4H)
  const float* Wh = (const float*)d_in[2];   // (H,4H)
  const float* b  = (const float*)d_in[3];   // (4H)
  const float* c0 = (const float*)d_in[4];   // (B,H)
  const float* h0 = (const float*)d_in[5];   // (B,H)
  const unsigned char* e0 = (const unsigned char*)d_in[6]; // (B) bool-ish (zeros)
  float* out = (float*)d_out;                // (T,B,H)

  char* ws = (char*)d_ws;
  size_t off = 0;
  __bf16* xp  = (__bf16*)(ws + off); off += (size_t)T_STEPS * BATCH * VDIM * 2; // 32MB
  __bf16* wip = (__bf16*)(ws + off); off += (size_t)VDIM * NGATE * 2;           // 16MB
  __bf16* whp = (__bf16*)(ws + off); off += (size_t)HDIM * NGATE * 2;           //  8MB
  float*  gx  = (float*) (ws + off); off += (size_t)T_STEPS * BATCH * NGATE * 4;//128MB
  __bf16* hp  = (__bf16*)(ws + off); off += (size_t)2 * BATCH * HDIM * 2;       //256KB
  float*  eos = (float*) (ws + off); off += (size_t)T_STEPS * BATCH * 4;        // 32KB
  int*    bar = (int*)   (ws + off); off += 256;

  hipMemsetAsync(bar, 0, 8, stream);

  { int n = T_STEPS * BATCH * VDIM;
    pack_a_kernel<<<(n + 255) / 256, 256, 0, stream>>>(x, xp, VDIM, VDIM / 32, n); }
  { int n = VDIM * NGATE;
    pack_b_kernel<<<(n + 255) / 256, 256, 0, stream>>>(Wi, wip, NGATE, VDIM / 32, n); }
  { int n = HDIM * NGATE;
    pack_b_kernel<<<(n + 255) / 256, 256, 0, stream>>>(Wh, whp, NGATE, HDIM / 32, n); }
  { int n = BATCH * HDIM;   // pack h0 into ping-pong buffer 0
    pack_a_kernel<<<(n + 255) / 256, 256, 0, stream>>>(h0, hp, HDIM, HDIM / 32, n); }

  eos_scan_kernel<<<1, 64, 0, stream>>>(x, e0, eos);

  gemm_bias_kernel<<<dim3(T_STEPS * BATCH / 64, NGATE / 64), 256, 0, stream>>>(
      xp, wip, b, gx, VDIM / 32, NGATE);

  lstm_rec_kernel<<<NBLK, 256, 0, stream>>>(gx, whp, hp, eos, c0, h0, out, bar);
}
